// MambaBlock_14654428414804
// MI455X (gfx1250) — compile-verified, matrix operand order checked by reading, not run on hardware
//
#include <hip/hip_runtime.h>
#include <hip/hip_bf16.h>

#define D_MODEL 1024
#define D_STATE 16
#define D_CONV  4
#define D_INNER 2048
#define BATCH   2
#define SEQLEN  1024
#define BL      (BATCH * SEQLEN)     // 2048 rows
#define N_XPROJ (1 + 2 * D_STATE)    // 33
#define EPS     1e-5f

typedef __attribute__((ext_vector_type(16))) __bf16 v16bf;
typedef __attribute__((ext_vector_type(8)))  __bf16 v8bf;
typedef __attribute__((ext_vector_type(8)))  float  v8f;

// ---------------------------------------------------------------------------
// 1. LayerNorm (fp32 in) -> bf16 out, one block per row
// ---------------------------------------------------------------------------
__global__ __launch_bounds__(256) void k_layernorm_bf16(
    const float* __restrict__ x, const float* __restrict__ g,
    const float* __restrict__ b, __bf16* __restrict__ out) {
  const int row = blockIdx.x;
  const float* xr = x + (size_t)row * D_MODEL;
  __shared__ float red[256];
  const int tid = threadIdx.x;

  float s = 0.f;
  for (int i = tid; i < D_MODEL; i += 256) s += xr[i];
  red[tid] = s; __syncthreads();
  for (int o = 128; o > 0; o >>= 1) {
    if (tid < o) red[tid] += red[tid + o];
    __syncthreads();
  }
  const float mu = red[0] * (1.f / D_MODEL);
  __syncthreads();

  float v = 0.f;
  for (int i = tid; i < D_MODEL; i += 256) { float d = xr[i] - mu; v += d * d; }
  red[tid] = v; __syncthreads();
  for (int o = 128; o > 0; o >>= 1) {
    if (tid < o) red[tid] += red[tid + o];
    __syncthreads();
  }
  const float rstd = rsqrtf(red[0] * (1.f / D_MODEL) + EPS);

  __bf16* orow = out + (size_t)row * D_MODEL;
  for (int i = tid; i < D_MODEL; i += 256)
    orow[i] = (__bf16)((xr[i] - mu) * rstd * g[i] + b[i]);
}

// ---------------------------------------------------------------------------
// 2. Transpose fp32 [R x C] -> bf16 [C x R] (LDS-tiled, coalesced both sides)
// ---------------------------------------------------------------------------
__global__ __launch_bounds__(256) void k_transpose_bf16(
    const float* __restrict__ in, __bf16* __restrict__ out, int R, int C) {
  __shared__ float tile[32][33];
  const int c0 = blockIdx.x * 32, r0 = blockIdx.y * 32;
  const int tx = threadIdx.x, ty = threadIdx.y;  // 32 x 8
  for (int i = ty; i < 32; i += 8) {
    int r = r0 + i, c = c0 + tx;
    tile[i][tx] = (r < R && c < C) ? in[(size_t)r * C + c] : 0.f;
  }
  __syncthreads();
  for (int i = ty; i < 32; i += 8) {
    int c = c0 + i, r = r0 + tx;
    if (c < C && r < R) out[(size_t)c * R + r] = (__bf16)tile[tx][i];
  }
}

// ---------------------------------------------------------------------------
// 3. WMMA bf16 GEMM:  C[M x N](f32) = A[M x K](bf16) * Bt[N x K](bf16)^T
//    32(M) x 64(N) tile per wave: 2 A-frags x 4 B-frags -> 8 WMMA / K-chunk.
//    A-frag (16x32): lane r(+16h) holds row r, K = {k0+8h..+7, k0+16+8h..+7}
//    D-frag: vgpr v on lane r(+16h) holds element (m = v + 8h, n = r)
// ---------------------------------------------------------------------------
__device__ __forceinline__ v16bf load_frag(const __bf16* p) {
  v8bf lo = *(const v8bf*)p;
  v8bf hi = *(const v8bf*)(p + 16);
  return __builtin_shufflevector(lo, hi, 0,1,2,3,4,5,6,7,8,9,10,11,12,13,14,15);
}

__global__ __launch_bounds__(256) void k_gemm_bf16_wmma(
    const __bf16* __restrict__ A, const __bf16* __restrict__ Bt,
    const float* __restrict__ resid, float* __restrict__ C,
    int M, int N, int K) {
  const int wave = threadIdx.x >> 5;
  const int lane = threadIdx.x & 31;
  const int wtile = blockIdx.x * 8 + wave;
  const int wn = N >> 6;                         // 64-col strips
  const int wtotal = (M >> 5) * wn;
  if (wtile >= wtotal) return;                   // wave-uniform: EXEC stays all-1
  const int tm = (wtile / wn) * 32;
  const int tn = (wtile % wn) * 64;

  const int half = lane >> 4;                    // 0 or 1
  const int r    = lane & 15;

  const __bf16* ap = A  + (size_t)(tm + r) * K + half * 8;
  const __bf16* bp = Bt + (size_t)(tn + r) * K + half * 8;
  const size_t strideA = (size_t)16 * K;         // next 16-row group
  const size_t strideB = (size_t)16 * K;         // next 16-col group

  v8f acc[2][4];
  #pragma unroll
  for (int i = 0; i < 2; ++i)
    #pragma unroll
    for (int j = 0; j < 4; ++j) acc[i][j] = (v8f){};

  for (int k0 = 0; k0 < K; k0 += 32) {
    if (k0 + 32 < K) {                           // -> global_prefetch_b8
      __builtin_prefetch((const void*)(ap + k0 + 32), 0, 3);
      __builtin_prefetch((const void*)(bp + k0 + 32), 0, 3);
    }
    v16bf af[2], bf[4];
    #pragma unroll
    for (int i = 0; i < 2; ++i) af[i] = load_frag(ap + strideA * i + k0);
    #pragma unroll
    for (int j = 0; j < 4; ++j) bf[j] = load_frag(bp + strideB * j + k0);
    #pragma unroll
    for (int i = 0; i < 2; ++i)
      #pragma unroll
      for (int j = 0; j < 4; ++j)
        acc[i][j] = __builtin_amdgcn_wmma_f32_16x16x32_bf16(
            false, af[i], false, bf[j], (short)0, acc[i][j], false, false);
  }

  #pragma unroll
  for (int i = 0; i < 2; ++i) {
    #pragma unroll
    for (int j = 0; j < 4; ++j) {
      const int ncol = tn + j * 16 + r;
      #pragma unroll
      for (int v = 0; v < 8; ++v) {
        const size_t idx = (size_t)(tm + i * 16 + half * 8 + v) * N + ncol;
        float val = acc[i][j][v];
        if (resid) val += resid[idx];
        C[idx] = val;
      }
    }
  }
}

// ---------------------------------------------------------------------------
// 4. Depthwise causal conv (k=4) + SiLU.  x_i lives in xz[:, 0:D_INNER].
// ---------------------------------------------------------------------------
__global__ __launch_bounds__(256) void k_conv_silu(
    const float* __restrict__ xz, const float* __restrict__ conv_w,
    const float* __restrict__ conv_b, float* __restrict__ xc) {
  const int idx = blockIdx.x * 256 + threadIdx.x;    // over BL * D_INNER
  if (idx >= BL * D_INNER) return;
  const int d   = idx & (D_INNER - 1);
  const int row = idx >> 11;                          // / D_INNER
  const int l   = row & (SEQLEN - 1);
  float acc = conv_b[d];
  #pragma unroll
  for (int j = 0; j < D_CONV; ++j) {
    const int ls = l + j - (D_CONV - 1);
    if (ls >= 0)
      acc += conv_w[d * D_CONV + j] *
             xz[(size_t)(row + j - (D_CONV - 1)) * (2 * D_INNER) + d];
  }
  xc[(size_t)row * D_INNER + d] = acc / (1.f + expf(-acc));
}

// ---------------------------------------------------------------------------
// 5. Small GEMM: bcdt[BL x 33] = xc[BL x D_INNER] * W_x[D_INNER x 33]
// ---------------------------------------------------------------------------
__global__ __launch_bounds__(256) void k_gemm_wx(
    const float* __restrict__ xc, const float* __restrict__ W_x,
    float* __restrict__ bcdt) {
  const int idx = blockIdx.x * 256 + threadIdx.x;
  if (idx >= BL * N_XPROJ) return;
  const int row = idx / N_XPROJ;
  const int e   = idx - row * N_XPROJ;
  const float* xr = xc + (size_t)row * D_INNER;
  float acc = 0.f;
  for (int d = 0; d < D_INNER; ++d) acc += xr[d] * W_x[d * N_XPROJ + e];
  bcdt[idx] = acc;
}

// ---------------------------------------------------------------------------
// 6. Selective scan, fused gating; writes y as bf16 for GEMM3.
//    1 thread per (b, d); h[16] in registers.
//    Per-timestep bcdt row (33 f32) staged into LDS with async copy:
//    9 lanes issue GLOBAL_LOAD_ASYNC_TO_LDS_B128, wave waits on ASYNCcnt,
//    then a workgroup barrier publishes the row.
// ---------------------------------------------------------------------------
__global__ __launch_bounds__(256) void k_scan(
    const float* __restrict__ bcdt, const float* __restrict__ xc,
    const float* __restrict__ xz,   const float* __restrict__ w_dt,
    const float* __restrict__ b_dt, const float* __restrict__ A_log,
    const float* __restrict__ D_param, __bf16* __restrict__ ybf) {
  const int b = blockIdx.x >> 3;                       // D_INNER/256 = 8 blocks/batch
  const int d = ((blockIdx.x & 7) << 8) + threadIdx.x;
  const float wd = w_dt[d], bd = b_dt[d], Dp = D_param[d];
  float Av[D_STATE];
  #pragma unroll
  for (int s = 0; s < D_STATE; ++s) Av[s] = -expf(A_log[d * D_STATE + s]);
  float h[D_STATE];
  #pragma unroll
  for (int s = 0; s < D_STATE; ++s) h[s] = 0.f;

  __shared__ float sb[36];                             // 33 used, b128-rounded
  // LDS aperture maps flat addr[31:0] -> wave-relative LDS byte address
  const unsigned lds_base = (unsigned)(uintptr_t)&sb[0];

  for (int l = 0; l < SEQLEN; ++l) {
    const int row = b * SEQLEN + l;
    if (threadIdx.x < 9) {                             // 9 x 16B = 144B >= 33 f32
      const float* gsrc = bcdt + row * N_XPROJ + threadIdx.x * 4;
      const unsigned ldst = lds_base + threadIdx.x * 16;
      asm volatile("global_load_async_to_lds_b128 %0, %1, off"
                   :: "v"(ldst), "v"(gsrc) : "memory");
    }
    asm volatile("s_wait_asynccnt 0x0" ::: "memory");
    __syncthreads();

    const float dtr = sb[0] * wd + bd;
    const float dt  = (dtr > 20.f) ? dtr : log1pf(expf(dtr));
    const float xcv = xc[(size_t)row * D_INNER + d];
    const float zv  = xz[(size_t)row * (2 * D_INNER) + D_INNER + d];
    float y = 0.f;
    #pragma unroll
    for (int s = 0; s < D_STATE; ++s) {
      const float dA = expf(dt * Av[s]);
      h[s] = dA * h[s] + dt * sb[1 + s] * xcv;
      y += h[s] * sb[1 + D_STATE + s];
    }
    y += xcv * Dp;
    const float gate = zv / (1.f + expf(-zv));
    ybf[(size_t)row * D_INNER + d] = (__bf16)(y * gate);
    __syncthreads();
  }
}

// ---------------------------------------------------------------------------
// launch
// ---------------------------------------------------------------------------
extern "C" void kernel_launch(void* const* d_in, const int* in_sizes, int n_in,
                              void* d_out, int out_size, void* d_ws, size_t ws_size,
                              hipStream_t stream) {
  const float* x      = (const float*)d_in[0];
  const float* ln_g   = (const float*)d_in[1];
  const float* ln_b   = (const float*)d_in[2];
  const float* W_in   = (const float*)d_in[3];
  const float* conv_w = (const float*)d_in[4];
  const float* conv_b = (const float*)d_in[5];
  const float* W_x    = (const float*)d_in[6];
  const float* w_dt   = (const float*)d_in[7];
  const float* b_dt   = (const float*)d_in[8];
  const float* A_log  = (const float*)d_in[9];
  const float* D_par  = (const float*)d_in[10];
  const float* W_out  = (const float*)d_in[11];
  float* out = (float*)d_out;

  char* ws = (char*)d_ws;
  size_t off = 0;
  __bf16* xn_bf  = (__bf16*)(ws + off); off += (size_t)BL * D_MODEL * 2;          // 4 MB
  __bf16* Wint   = (__bf16*)(ws + off); off += (size_t)(2*D_INNER) * D_MODEL * 2; // 8 MB
  float*  xz     = (float*) (ws + off); off += (size_t)BL * 2*D_INNER * 4;        // 32 MB
  float*  xc     = (float*) (ws + off); off += (size_t)BL * D_INNER * 4;          // 16 MB
  float*  bcdt   = (float*) (ws + off); off += (size_t)BL * N_XPROJ * 4;          // .26 MB
  __bf16* ybf    = (__bf16*)(ws + off); off += (size_t)BL * D_INNER * 2;          // 8 MB
  __bf16* Woutt  = (__bf16*)(ws + off); off += (size_t)D_MODEL * D_INNER * 2;     // 4 MB

  // 1. LayerNorm -> bf16
  k_layernorm_bf16<<<BL, 256, 0, stream>>>(x, ln_g, ln_b, xn_bf);

  // 2. Weight transpose+convert: W_in [1024 x 4096] -> [4096 x 1024] bf16
  k_transpose_bf16<<<dim3(2*D_INNER/32, D_MODEL/32), dim3(32, 8), 0, stream>>>(
      W_in, Wint, D_MODEL, 2*D_INNER);
  //    W_out [2048 x 1024] -> [1024 x 2048] bf16
  k_transpose_bf16<<<dim3(D_MODEL/32, D_INNER/32), dim3(32, 8), 0, stream>>>(
      W_out, Woutt, D_INNER, D_MODEL);

  // 3. GEMM1: xz = xn @ W_in   (M=2048, N=4096, K=1024)
  //    waves = (M/32)*(N/64) = 4096 -> 512 blocks of 8 waves
  k_gemm_bf16_wmma<<<(BL/32) * (2*D_INNER/64) / 8, 256, 0, stream>>>(
      xn_bf, Wint, nullptr, xz, BL, 2*D_INNER, D_MODEL);

  // 4. causal depthwise conv + SiLU
  k_conv_silu<<<(BL * D_INNER) / 256, 256, 0, stream>>>(xz, conv_w, conv_b, xc);

  // 5. bcdt = xc @ W_x
  k_gemm_wx<<<(BL * N_XPROJ + 255) / 256, 256, 0, stream>>>(xc, W_x, bcdt);

  // 6. selective scan + gating -> ybf
  k_scan<<<BATCH * (D_INNER / 256), 256, 0, stream>>>(
      bcdt, xc, xz, w_dt, b_dt, A_log, D_par, ybf);

  // 7. GEMM3: out = y @ W_out + x   (M=2048, N=1024, K=2048)
  //    waves = (2048/32)*(1024/64) = 1024 -> 128 blocks
  k_gemm_bf16_wmma<<<(BL/32) * (D_MODEL/64) / 8, 256, 0, stream>>>(
      ybf, Woutt, x, out, BL, D_MODEL, D_INNER);
}
